// GCAT_41927470744111
// MI455X (gfx1250) — matrix operand order repaired and checked by reading, not compiled, not run on hardware
//
#include <hip/hip_runtime.h>
#include <hip/hip_bf16.h>
#include <math.h>

// ---------------- problem constants (from reference) ----------------
#define NN      50000   // nodes
#define DEG     16      // edges per node (rows = repeat(arange(N),16) -> w = 1/16)
#define EE      800000  // edges
#define RR      1000    // relations
#define AA      2000    // attributes
#define ADEG    4       // attrs per node -> inv_adeg = 1/4
#define DOUT    128
#define HH      4
#define DH      32
#define LL      2
#define FD      192     // feature dim of F = [e(128) | concept_rel(32) | concept_attr(32)]
#define NTILES  (NN / 16)        // 3125 (exact)
#define EBLK    (EE / 256)       // 3125 edge blocks (exact)

typedef _Float16 v16h __attribute__((ext_vector_type(16)));
typedef _Float16 v8h  __attribute__((ext_vector_type(8)));
typedef float    v8f  __attribute__((ext_vector_type(8)));

// ---------------- weight assembly: W[d][col] for S = F @ W ----------------
// col 0..3 = s_self head h, col 4..7 = s_neigh head h, col 8..15 = zero pad.
// d in [32h,32h+32): self -> k[d-32h]      (k[:o0]),   neigh -> k[192+d-32h] (k[o5:])
// d in [128,160):    self -> k[d-96]       (k[o0:o1]), neigh -> k[d]         (k[o3:o4])
// d in [160,192):    self -> k[d-96]       (k[o1:o2]), neigh -> k[d]         (k[o4:o5])
__device__ inline float wval(const float* attn, int l, int d, int colN) {
    if (colN >= 8) return 0.0f;
    int  h      = colN & 3;
    bool selfcl = colN < 4;
    const float* k = attn + (size_t)(l * HH + h) * 224;
    if (d < 128) {
        if ((d >> 5) == h) return selfcl ? k[d - 32 * h] : k[192 + d - 32 * h];
        return 0.0f;
    }
    return selfcl ? k[d - 96] : k[d];
}

// Pack B into exact WMMA B-operand per-lane layout:
// lanes 0-15 hold K = 32c+0..7,16..23 (elem j<8 -> K=32c+j, j>=8 -> K=32c+8+j, N=lane),
// lanes 16-31 hold K = 32c+8..15,24..31 (elem j<8 -> K=32c+8+j, j>=8 -> K=32c+16+j, N=lane-16).
__global__ void prep_bpack(const float* __restrict__ attn, _Float16* __restrict__ Bpack) {
    int idx = blockIdx.x * blockDim.x + threadIdx.x;
    if (idx >= LL * 6 * 32) return;
    int l    = idx / (6 * 32);
    int rem  = idx % (6 * 32);
    int c    = rem >> 5;
    int lane = rem & 31;
    int kb   = c * 32 + ((lane >> 4) << 3);
    int ncol = lane & 15;
    for (int j = 0; j < 16; ++j)
        Bpack[(size_t)idx * 16 + j] = (_Float16)wval(attn, l, kb + (j < 8 ? j : j + 8), ncol);
}

// s_rel[l][r][h] = rel_emb[r] . k[l,h,96:128]
__global__ void srel_kernel(const float* __restrict__ relE, const float* __restrict__ attn,
                            float* __restrict__ Srel) {
    int idx = blockIdx.x * blockDim.x + threadIdx.x;
    if (idx >= LL * RR * HH) return;
    int h = idx & 3;
    int r = (idx >> 2) % RR;
    int l = (idx >> 2) / RR;
    const float* k  = attn + (size_t)(l * HH + h) * 224 + 96;
    const float* re = relE + (size_t)r * 32;
    float s = 0.0f;
    for (int i = 0; i < 32; ++i) s += re[i] * k[i];
    Srel[idx] = s;
}

// Per node (one wave32 each): x0 = mean ent_emb[col], cr = relu(mean rel_emb),
// ca = relu(mean attr_emb). float4 loads: 32 lanes * 16B cover a 512B row.
__global__ void precompute_kernel(const float* __restrict__ ent, const float* __restrict__ relE,
                                  const float* __restrict__ attrE, const int* __restrict__ eidx,
                                  const int* __restrict__ erel, const int* __restrict__ aidx,
                                  float* __restrict__ x0, float* __restrict__ cr,
                                  float* __restrict__ ca) {
    int lane = threadIdx.x & 31;
    int n    = (int)((blockIdx.x * blockDim.x + threadIdx.x) >> 5);
    if (n >= NN) return;
    size_t eb = (size_t)n * DEG;
    float4 ax = make_float4(0.f, 0.f, 0.f, 0.f);
    float  ar = 0.0f;
    for (int j = 0; j < DEG; ++j) {
        int col = eidx[(eb + j) * 2 + 1];
        float4 v = *((const float4*)(ent + (size_t)col * DOUT) + lane);
        ax.x += v.x; ax.y += v.y; ax.z += v.z; ax.w += v.w;
        ar += relE[(size_t)erel[eb + j] * 32 + lane];
    }
    const float w = 1.0f / 16.0f;
    *((float4*)(x0 + (size_t)n * DOUT) + lane) =
        make_float4(ax.x * w, ax.y * w, ax.z * w, ax.w * w);
    cr[(size_t)n * 32 + lane] = fmaxf(ar * w, 0.0f);
    float aa = 0.0f;
    for (int j = 0; j < ADEG; ++j)
        aa += attrE[(size_t)aidx[((size_t)n * ADEG + j) * 2 + 1] * 32 + lane];
    ca[(size_t)n * 32 + lane] = fmaxf(aa * 0.25f, 0.0f);
}

// F[n][0:128] = relu(x[n]) (f16), F[n][128:160] = cr, F[n][160:192] = ca
__global__ void build_F(const float* __restrict__ x, int xstride, const float* __restrict__ cr,
                        const float* __restrict__ ca, _Float16* __restrict__ F) {
    size_t idx = (size_t)blockIdx.x * blockDim.x + threadIdx.x;
    if (idx >= (size_t)NN * FD) return;
    int n = (int)(idx / FD);
    int d = (int)(idx % FD);
    float v;
    if (d < 128)      v = fmaxf(x[(size_t)n * xstride + d], 0.0f);
    else if (d < 160) v = cr[(size_t)n * 32 + (d - 128)];
    else              v = ca[(size_t)n * 32 + (d - 160)];
    F[idx] = (_Float16)v;
}

// S (N x 8) = F (N x 192) @ W (192 x 8) via v_wmma_f32_16x16x32_f16.
// One wave per 16-node tile, 6 K-chunks of 32.
__global__ void score_wmma(const _Float16* __restrict__ F, const _Float16* __restrict__ Bp,
                           float* __restrict__ S) {
    int lane = threadIdx.x & 31;
    int tile = (int)((blockIdx.x * blockDim.x + threadIdx.x) >> 5);
    if (tile >= NTILES) return;
    int nb  = tile * 16;
    int row = nb + (lane & 15);
    int sel = (lane >> 4) * 8;     // lanes>=16 hold K+8..15 / K+24..31 of each chunk
    const _Float16* Fr = F + (size_t)row * FD + sel;
    v8f acc = {};
    #pragma unroll
    for (int c = 0; c < 6; ++c) {
        v8h lo = *(const v8h*)(Fr + c * 32);        // K offsets c*32 + sel + [0..7]
        v8h hi = *(const v8h*)(Fr + c * 32 + 16);   // K offsets c*32 + sel + [16..23]
        v16h a;
        #pragma unroll
        for (int i = 0; i < 8; ++i) { a[i] = lo[i]; a[i + 8] = hi[i]; }
        v16h b = *(const v16h*)(Bp + ((size_t)c * 32 + lane) * 16);
        acc = __builtin_amdgcn_wmma_f32_16x16x32_f16(false, a, false, b, (short)0, acc,
                                                     false, false);
    }
    // D layout: VGPR r -> M=r (lanes 0-15) / M=8+r (lanes 16-31), N = lane&15
    int ncol = lane & 15;
    if (ncol < 8) {
        int mb = (lane < 16) ? 0 : 8;
        #pragma unroll
        for (int r = 0; r < 8; ++r)
            S[(size_t)(nb + mb + r) * 8 + ncol] = acc[r];
    }
}

// s = leaky_relu(S_self[row,h] + S_rel[rel,h] + S_neigh[col,h], 0.3); track block max/head.
__global__ void edge_scores(const float* __restrict__ S, const float* __restrict__ Srel,
                            const int* __restrict__ eidx, const int* __restrict__ erel,
                            float* __restrict__ sE, float* __restrict__ pmax) {
    int    tid = threadIdx.x;
    size_t e   = (size_t)blockIdx.x * 256 + tid;
    float  lm[HH];
    #pragma unroll
    for (int h = 0; h < HH; ++h) lm[h] = -INFINITY;
    if (e < EE) {
        int n   = (int)(e >> 4);
        int col = eidx[e * 2 + 1];
        int r   = erel[e];
        #pragma unroll
        for (int h = 0; h < HH; ++h) {
            float s = S[(size_t)n * 8 + h] + Srel[(size_t)r * 4 + h] + S[(size_t)col * 8 + 4 + h];
            s = (s > 0.0f) ? s : 0.3f * s;
            sE[e * 4 + h] = s;
            lm[h] = s;
        }
    }
    __shared__ float sm[HH * 256];
    #pragma unroll
    for (int h = 0; h < HH; ++h) sm[h * 256 + tid] = lm[h];
    __syncthreads();
    for (int st = 128; st > 0; st >>= 1) {
        if (tid < st)
            for (int h = 0; h < HH; ++h)
                sm[h * 256 + tid] = fmaxf(sm[h * 256 + tid], sm[h * 256 + tid + st]);
        __syncthreads();
    }
    if (tid < HH) pmax[(size_t)blockIdx.x * HH + tid] = sm[tid * 256];
}

// partial sums of exp(s - gmax[h]) per head
__global__ void exp_sums(const float* __restrict__ sE, const float* __restrict__ gmax,
                         float* __restrict__ psum) {
    int    tid = threadIdx.x;
    size_t e   = (size_t)blockIdx.x * 256 + tid;
    float  ls[HH];
    #pragma unroll
    for (int h = 0; h < HH; ++h) ls[h] = 0.0f;
    if (e < EE) {
        #pragma unroll
        for (int h = 0; h < HH; ++h) ls[h] = expf(sE[e * 4 + h] - gmax[h]);
    }
    __shared__ float sm[HH * 256];
    #pragma unroll
    for (int h = 0; h < HH; ++h) sm[h * 256 + tid] = ls[h];
    __syncthreads();
    for (int st = 128; st > 0; st >>= 1) {
        if (tid < st)
            for (int h = 0; h < HH; ++h)
                sm[h * 256 + tid] += sm[h * 256 + tid + st];
        __syncthreads();
    }
    if (tid < HH) psum[(size_t)blockIdx.x * HH + tid] = sm[tid * 256];
}

// single-block reduction of [n][4] partials (max or sum)
__global__ void reduce4(const float* __restrict__ part, int n, float* __restrict__ outv,
                        int isMax) {
    int tid = threadIdx.x;
    float acc[HH];
    #pragma unroll
    for (int h = 0; h < HH; ++h) acc[h] = isMax ? -INFINITY : 0.0f;
    for (int i = tid; i < n; i += 256)
        for (int h = 0; h < HH; ++h) {
            float v = part[(size_t)i * HH + h];
            acc[h] = isMax ? fmaxf(acc[h], v) : (acc[h] + v);
        }
    __shared__ float sm[HH * 256];
    #pragma unroll
    for (int h = 0; h < HH; ++h) sm[h * 256 + tid] = acc[h];
    __syncthreads();
    for (int st = 128; st > 0; st >>= 1) {
        if (tid < st)
            for (int h = 0; h < HH; ++h) {
                float a = sm[h * 256 + tid], b = sm[h * 256 + tid + st];
                sm[h * 256 + tid] = isMax ? fmaxf(a, b) : (a + b);
            }
        __syncthreads();
    }
    if (tid < HH) outv[tid] = sm[tid * 256];
}

// One wave per node: global softmax value p, per-row softmax a (lanes 0-15 = edges),
// then out[n, l*128 + h*32 + d] = tanh( sum_j a_j * e[col_j, h*32 + d] ), d = lane.
__global__ void aggregate(const _Float16* __restrict__ F, const int* __restrict__ eidx,
                          const float* __restrict__ sE, const float* __restrict__ gmax,
                          const float* __restrict__ gsum, float* __restrict__ outp, int layer) {
    int lane = threadIdx.x & 31;
    int n    = (int)((blockIdx.x * blockDim.x + threadIdx.x) >> 5);
    if (n >= NN) return;
    size_t eb = (size_t)n * DEG;
    int colr = 0;
    if (lane < 16) colr = eidx[(eb + lane) * 2 + 1];
    #pragma unroll
    for (int h = 0; h < HH; ++h) {
        float gm  = gmax[h];
        float gis = 1.0f / gsum[h];
        float p   = (lane < 16) ? expf(sE[(eb + lane) * 4 + h] - gm) * gis : -INFINITY;
        float m   = p;
        m = fmaxf(m, __shfl_xor(m, 16, 32));
        m = fmaxf(m, __shfl_xor(m, 8, 32));
        m = fmaxf(m, __shfl_xor(m, 4, 32));
        m = fmaxf(m, __shfl_xor(m, 2, 32));
        m = fmaxf(m, __shfl_xor(m, 1, 32));
        float z  = (lane < 16) ? expf(p - m) : 0.0f;
        float zs = z;
        zs += __shfl_xor(zs, 16, 32);
        zs += __shfl_xor(zs, 8, 32);
        zs += __shfl_xor(zs, 4, 32);
        zs += __shfl_xor(zs, 2, 32);
        zs += __shfl_xor(zs, 1, 32);
        float a = z / zs;
        float acc = 0.0f;
        #pragma unroll
        for (int j = 0; j < DEG; ++j) {
            float aj = __shfl(a, j, 32);
            int   cj = __shfl(colr, j, 32);
            acc += aj * (float)F[(size_t)cj * FD + h * DH + lane];
        }
        outp[(size_t)n * (LL * DOUT) + layer * DOUT + h * DH + lane] = tanhf(acc);
    }
}

// ---------------------------------------------------------------------------
extern "C" void kernel_launch(void* const* d_in, const int* in_sizes, int n_in,
                              void* d_out, int out_size, void* d_ws, size_t ws_size,
                              hipStream_t stream) {
    (void)in_sizes; (void)n_in; (void)out_size; (void)ws_size;
    const float* ent  = (const float*)d_in[0];   // (N,128)
    const float* relE = (const float*)d_in[1];   // (R,32)
    const float* attE = (const float*)d_in[2];   // (A,32)
    const float* attn = (const float*)d_in[3];   // (L,H,224)
    const int*   eidx = (const int*)d_in[4];     // (E,2)
    const int*   erel = (const int*)d_in[5];     // (E,)
    const int*   aidx = (const int*)d_in[6];     // (AE,2)
    float*       outp = (float*)d_out;           // (N,256)

    // workspace layout
    char*  ws  = (char*)d_ws;
    size_t off = 0;
    auto take = [&](size_t bytes) { char* p = ws + off; off = (off + bytes + 255) & ~(size_t)255; return p; };
    float*    x0    = (float*)take((size_t)NN * DOUT * 4);
    float*    cr    = (float*)take((size_t)NN * 32 * 4);
    float*    ca    = (float*)take((size_t)NN * 32 * 4);
    _Float16* F     = (_Float16*)take((size_t)NN * FD * 2);
    float*    S     = (float*)take((size_t)NN * 8 * 4);
    float*    Srel  = (float*)take((size_t)LL * RR * HH * 4);
    float*    sE    = (float*)take((size_t)EE * HH * 4);
    float*    pmax  = (float*)take((size_t)EBLK * HH * 4);
    float*    psum  = (float*)take((size_t)EBLK * HH * 4);
    float*    gmax  = (float*)take(HH * 4);
    float*    gsum  = (float*)take(HH * 4);
    _Float16* Bpack = (_Float16*)take((size_t)LL * 6 * 32 * 16 * 2);

    prep_bpack<<<2, 256, 0, stream>>>(attn, Bpack);
    srel_kernel<<<(LL * RR * HH + 255) / 256, 256, 0, stream>>>(relE, attn, Srel);
    precompute_kernel<<<NN / 8, 256, 0, stream>>>(ent, relE, attE, eidx, erel, aidx, x0, cr, ca);

    for (int l = 0; l < LL; ++l) {
        const float* xin = (l == 0) ? x0 : outp;      // layer 1 reads layer 0 output slice
        int xstride      = (l == 0) ? DOUT : LL * DOUT;
        build_F<<<(int)(((size_t)NN * FD + 255) / 256), 256, 0, stream>>>(xin, xstride, cr, ca, F);
        score_wmma<<<(NTILES + 7) / 8, 256, 0, stream>>>(F, Bpack + (size_t)l * 6 * 32 * 16, S);
        edge_scores<<<EBLK, 256, 0, stream>>>(S, Srel + (size_t)l * RR * HH, eidx, erel, sE, pmax);
        reduce4<<<1, 256, 0, stream>>>(pmax, EBLK, gmax, 1);
        exp_sums<<<EBLK, 256, 0, stream>>>(sE, gmax, psum);
        reduce4<<<1, 256, 0, stream>>>(psum, EBLK, gsum, 0);
        aggregate<<<NN / 8, 256, 0, stream>>>(F, eidx, sE, gmax, gsum, outp, l);
    }
}
